// VectorQuantizerEMA_37056977830102
// MI455X (gfx1250) — matrix operand order repaired and checked by reading, not compiled, not run on hardware
//
#include <hip/hip_runtime.h>
#include <math.h>

typedef __attribute__((ext_vector_type(16))) _Float16 v16h;
typedef __attribute__((ext_vector_type(8)))  _Float16 v8h;
typedef __attribute__((ext_vector_type(8)))  float    v8f;

typedef __attribute__((address_space(3))) _Float16 lds_f16_t;
typedef __attribute__((address_space(3))) float    lds_f32_t;

#define N_TOT 32768      // B*S
#define DIM   256        // D
#define NCODE 8192       // K
#define KT    64         // codes per LDS tile (per buffer)
#define LDS_STRIDE 264   // DIM + 8 f16 pad -> 528B rows: bank-conflict-free b128
#define ROWS_PER_BLOCK 128
#define THREADS 256
#define BETA 0.25f

// ---------------------------------------------------------------------------
// Prep: embed(f32) -> f16 copy, per-code squared norms, zero counts/loss.
// ---------------------------------------------------------------------------
__global__ __launch_bounds__(256) void vq_prep(const float* __restrict__ embed,
                                               _Float16* __restrict__ ef16,
                                               float* __restrict__ enorm,
                                               int* __restrict__ counts,
                                               float* __restrict__ loss_acc) {
    const int k = blockIdx.x;
    const int t = threadIdx.x;
    const float v = embed[(size_t)k * DIM + t];
    ef16[(size_t)k * DIM + t] = (_Float16)v;
    float s = v * v;
    #pragma unroll
    for (int m = 16; m >= 1; m >>= 1) s += __shfl_xor(s, m, 32);
    __shared__ float red[8];
    if ((t & 31) == 0) red[t >> 5] = s;
    __syncthreads();
    if (t == 0) {
        float tot = 0.f;
        #pragma unroll
        for (int w = 0; w < 8; ++w) tot += red[w];
        enorm[k]  = tot;
        counts[k] = 0;
        if (k == 0) *loss_acc = 0.f;
    }
}

// ---------------------------------------------------------------------------
// Main: fused distance-GEMM (f16 WMMA) + argmin + gather + loss partials.
// 8 waves/block, 16 rows/wave, 128 rows/block. Double-buffered code tiles
// staged with GLOBAL_LOAD_ASYNC_TO_LDS (ASYNCcnt), overlapped with WMMA.
// ---------------------------------------------------------------------------
__global__ __launch_bounds__(256) void vq_main(const float* __restrict__ z,
                                               const float* __restrict__ embed,
                                               const _Float16* __restrict__ ef16,
                                               const float* __restrict__ enorm,
                                               float* __restrict__ out_zq,
                                               float* __restrict__ out_idx_f,
                                               int* __restrict__ counts,
                                               float* __restrict__ loss_acc) {
    __shared__ __align__(64) _Float16 e_lds[2][KT][LDS_STRIDE];  // ~66 KB
    __shared__ __align__(16) float    n_lds[2][KT];
    __shared__ int   idx_lds[ROWS_PER_BLOCK];
    __shared__ float red_lds[8];

    const int tid  = threadIdx.x;
    const int lane = tid & 31;
    const int wave = tid >> 5;
    const int half = lane >> 4;    // which 16-lane half of the wave32
    const int l15  = lane & 15;

    const int row_base = blockIdx.x * ROWS_PER_BLOCK + wave * 16;
    const int m_row    = row_base + l15;              // this lane's A row

    // LDS byte offsets for async-copy destinations.
    const unsigned e_base0 = (unsigned)(unsigned long long)(lds_f16_t*)&e_lds[0][0][0];
    const unsigned e_base1 = (unsigned)(unsigned long long)(lds_f16_t*)&e_lds[1][0][0];
    const unsigned n_base0 = (unsigned)(unsigned long long)(lds_f32_t*)&n_lds[0][0];
    const unsigned n_base1 = (unsigned)(unsigned long long)(lds_f32_t*)&n_lds[1][0];

    // Stage one KT x DIM f16 tile (+ norms) into LDS buffer `nbuf` using
    // direct global->LDS async copies (no VGPR round trip, ASYNCcnt-tracked).
    auto stage_tile = [&](int kt_next, int nbuf) {
        const unsigned ebase = nbuf ? e_base1 : e_base0;
        #pragma unroll
        for (int v = 0; v < 8; ++v) {
            const int c   = tid + v * THREADS;       // 2048 x 16B chunks
            const int row = c >> 5;                  // 32 chunks per code row
            const int cc  = c & 31;
            const unsigned lds_off = ebase + (unsigned)(row * (LDS_STRIDE * 2) + cc * 16);
            const unsigned long long ga =
                (unsigned long long)(ef16 + (size_t)kt_next * DIM + row * DIM + cc * 8);
            asm volatile("global_load_async_to_lds_b128 %0, %1, off"
                         :: "v"(lds_off), "v"(ga) : "memory");
        }
        if (tid < KT) {
            const unsigned lds_off = (nbuf ? n_base1 : n_base0) + (unsigned)(tid * 4);
            const unsigned long long ga = (unsigned long long)(enorm + kt_next + tid);
            asm volatile("global_load_async_to_lds_b32 %0, %1, off"
                         :: "v"(lds_off), "v"(ga) : "memory");
        }
    };

    // ---- Preload A fragments (ISA 16-bit A 16x32 layout), reused over all K.
    v16h a[8];
    {
        const float* zrow = z + (size_t)m_row * DIM;
        #pragma unroll
        for (int kf = 0; kf < 8; ++kf) {
            const int k0 = kf * 32 + half * 8;
            #pragma unroll
            for (int i = 0; i < 8; ++i) {
                a[kf][i]     = (_Float16)zrow[k0 + i];
                a[kf][8 + i] = (_Float16)zrow[k0 + 16 + i];
            }
        }
    }

    float minval[8];
    int   minidx[8];
    #pragma unroll
    for (int r = 0; r < 8; ++r) { minval[r] = 3.4e38f; minidx[r] = 0; }

    // Prologue: stage tile 0 into buffer 0.
    stage_tile(0, 0);
    asm volatile("s_wait_asynccnt 0x0" ::: "memory");
    __syncthreads();

    int buf = 0;
    for (int kt = 0; kt < NCODE; kt += KT, buf ^= 1) {
        // Kick off the async copy of the next tile into the other buffer.
        if (kt + KT < NCODE) stage_tile(kt + KT, buf ^ 1);

        #pragma unroll
        for (int ng = 0; ng < KT / 16; ++ng) {
            const int nloc = ng * 16 + l15;       // this lane's B column
            v8f acc = {0.f, 0.f, 0.f, 0.f, 0.f, 0.f, 0.f, 0.f};
            #pragma unroll
            for (int kf = 0; kf < 8; ++kf) {
                // B (32x16) layout: lane holds col nloc, K runs [8h,8h+8) and [16+8h,24+8h).
                const _Float16* bp = &e_lds[buf][nloc][kf * 32 + half * 8];
                const v8h b0 = *(const v8h*)bp;
                const v8h b1 = *(const v8h*)(bp + 16);
                const v16h b = __builtin_shufflevector(
                    b0, b1, 0, 1, 2, 3, 4, 5, 6, 7, 8, 9, 10, 11, 12, 13, 14, 15);
                acc = __builtin_amdgcn_wmma_f32_16x16x32_f16(
                        false, a[kf], false, b, (short)0, acc, false, false);
            }
            const float en   = n_lds[buf][nloc];
            const int   cidx = kt + nloc;
            #pragma unroll
            for (int r = 0; r < 8; ++r) {
                const float s = en - 2.0f * acc[r];  // ||e||^2 - 2 z.e
                if (s < minval[r]) { minval[r] = s; minidx[r] = cidx; }
            }
        }

        // My async writes for the next tile are done, then publish to all waves.
        asm volatile("s_wait_asynccnt 0x0" ::: "memory");
        __syncthreads();
    }

    // ---- Cross-lane argmin over the 16 lanes of each half-wave (C: col=l15).
    #pragma unroll
    for (int r = 0; r < 8; ++r) {
        float v = minval[r];
        int   ix = minidx[r];
        #pragma unroll
        for (int msk = 1; msk <= 8; msk <<= 1) {
            const float ov = __shfl_xor(v, msk, 32);
            const int   oi = __shfl_xor(ix, msk, 32);
            if (ov < v || (ov == v && oi < ix)) { v = ov; ix = oi; }
        }
        minval[r] = v; minidx[r] = ix;
    }
    if (l15 == 0) {
        #pragma unroll
        for (int r = 0; r < 8; ++r) {
            const int row_loc = wave * 16 + half * 8 + r;  // C row = r + 8*half
            const int grow    = blockIdx.x * ROWS_PER_BLOCK + row_loc;
            idx_lds[row_loc] = minidx[r];
            out_idx_f[grow]  = (float)minidx[r];
            atomicAdd(&counts[minidx[r]], 1);
        }
    }
    __syncthreads();

    // ---- Gather z_q = embed[idx] (fp32, coalesced) + commitment-loss partial.
    float lsum = 0.f;
    const size_t base = (size_t)blockIdx.x * ROWS_PER_BLOCK * DIM;
    for (int i = tid; i < ROWS_PER_BLOCK * DIM; i += THREADS) {
        const int row = i >> 8;           // DIM == 256
        const int col = i & 255;
        const int ci  = idx_lds[row];
        const float q  = embed[(size_t)ci * DIM + col];
        const float zz = z[base + i];
        out_zq[base + i] = q;             // z + sg(z_q - z) == z_q numerically
        const float df = q - zz;
        lsum += df * df;
    }
    #pragma unroll
    for (int msk = 16; msk >= 1; msk >>= 1) lsum += __shfl_xor(lsum, msk, 32);
    if (lane == 0) red_lds[wave] = lsum;
    __syncthreads();
    if (tid == 0) {
        float t = 0.f;
        #pragma unroll
        for (int w = 0; w < 8; ++w) t += red_lds[w];
        atomicAdd(loss_acc, t);
    }
}

// ---------------------------------------------------------------------------
// Finalize: loss scalar + perplexity from the code-usage histogram.
// ---------------------------------------------------------------------------
__global__ __launch_bounds__(256) void vq_fin(const int* __restrict__ counts,
                                              const float* __restrict__ loss_acc,
                                              float* __restrict__ out_scal) {
    const int t = threadIdx.x;
    float H = 0.f;
    for (int k = t; k < NCODE; k += 256) {
        const float p = (float)counts[k] * (1.0f / (float)N_TOT);
        H += p * logf(p + 1e-12f);
    }
    #pragma unroll
    for (int m = 16; m >= 1; m >>= 1) H += __shfl_xor(H, m, 32);
    __shared__ float red[8];
    if ((t & 31) == 0) red[t >> 5] = H;
    __syncthreads();
    if (t == 0) {
        float h = 0.f;
        #pragma unroll
        for (int w = 0; w < 8; ++w) h += red[w];
        out_scal[0] = BETA * (*loss_acc) / (float)(N_TOT * DIM);  // loss
        out_scal[1] = expf(-h);                                   // perplexity
    }
}

// ---------------------------------------------------------------------------
extern "C" void kernel_launch(void* const* d_in, const int* in_sizes, int n_in,
                              void* d_out, int out_size, void* d_ws, size_t ws_size,
                              hipStream_t stream) {
    const float* z     = (const float*)d_in[0];   // [B,S,D] f32
    const float* embed = (const float*)d_in[1];   // [K,D]   f32

    char* ws = (char*)d_ws;
    _Float16* ef16  = (_Float16*)ws;                            // 4 MB
    float*    enorm = (float*)(ws + (size_t)NCODE * DIM * 2);   // 32 KB
    int*      cnts  = (int*)(ws + (size_t)NCODE * DIM * 2 + (size_t)NCODE * 4);
    float*    lacc  = (float*)(ws + (size_t)NCODE * DIM * 2 + (size_t)NCODE * 8);

    float* out       = (float*)d_out;
    float* out_zq    = out;                                 // N*D
    float* out_scal  = out + (size_t)N_TOT * DIM;           // loss, perplexity
    float* out_idx_f = out + (size_t)N_TOT * DIM + 2;       // N indices (as f32)

    vq_prep<<<NCODE, 256, 0, stream>>>(embed, ef16, enorm, cnts, lacc);
    vq_main<<<N_TOT / ROWS_PER_BLOCK, THREADS, 0, stream>>>(
        z, embed, ef16, enorm, out_zq, out_idx_f, cnts, lacc);
    vq_fin<<<1, 256, 0, stream>>>(cnts, lacc, out_scal);
}
